// RWKVTimeMixingChunked_8675833938459
// MI455X (gfx1250) — matrix operand order repaired and checked by reading, not compiled, not run on hardware
//
#include <hip/hip_runtime.h>

typedef __attribute__((ext_vector_type(16))) _Float16 v16h;
typedef __attribute__((ext_vector_type(8)))  float    v8f;

#define D_ATTN  1024
#define D_STATE 64
#define TLEN    8192
#define BATCH   4
#define CHUNKL  128
#define CPB     (TLEN/CHUNKL)       /* 64 chunks per batch  */
#define NCHUNK  (BATCH*CPB)         /* 256 total chunks     */
#define MROWS   (BATCH*TLEN)        /* 32768                */

#define AK   32
#define ASTR 36                     /* 32 halves + 4 pad per LDS row */
#define TSZ  (128*ASTR)             /* halves per LDS tile  */

union HFrag { v16h h; unsigned u[8]; };

__device__ inline float fast_sigmoid(float x) {
  return __builtin_amdgcn_rcpf(1.f + __expf(-x));
}

// ---- WMMA fragment loaders (ISA 7.12.2 layouts, wave32) -------------------
// A (16x32 f16): lanes 0-15 = rows 0-15 with K pairs {0..7,16..23};
//                lanes 16-31 = same rows with K pairs {8..15,24..31}.
__device__ inline v16h frag_a(const _Float16* base, int lane) {
  const int m = lane & 15, hi = lane >> 4;
  const unsigned* rp = (const unsigned*)(base + m * ASTR);
  HFrag f;
#pragma unroll
  for (int j = 0; j < 8; ++j) {
    const int p = (j < 4 ? j : j + 4) + 4 * hi;   // dword index; K = 2p,2p+1
    f.u[j] = rp[p];
  }
  return f.h;
}
// B (32x16 f16), LDS staged transposed as [n][k]: lanes 0-15 = cols, K {0..7,16..23};
// lanes 16-31 = same cols, K {8..15,24..31} (matches A).
__device__ inline v16h frag_b(const _Float16* base, int lane) {
  const int n = lane & 15, hi = lane >> 4;
  const unsigned* cp = (const unsigned*)(base + n * ASTR);
  HFrag f;
#pragma unroll
  for (int j = 0; j < 8; ++j) f.u[j] = cp[(j < 4 ? j : j + 4) + 4 * hi];
  return f.h;
}

// ---- shared GEMM helpers: 128x128 block, 8 waves (2x4), 64x32 per wave ----
__device__ inline void store_a_tile(_Float16* As, int tid, const float4 av[4]) {
  const int rl = tid >> 3, kc = (tid & 7) * 4;
#pragma unroll
  for (int r4 = 0; r4 < 4; ++r4) {
    _Float16* dst = As + (r4 * 32 + rl) * ASTR + kc;
    dst[0] = (_Float16)av[r4].x; dst[1] = (_Float16)av[r4].y;
    dst[2] = (_Float16)av[r4].z; dst[3] = (_Float16)av[r4].w;
  }
}
__device__ inline void store_b_tile(_Float16* Bs, int tid, const float2 bv[8]) {
  const int k0 = tid >> 6, n = (tid & 63) * 2;
#pragma unroll
  for (int kk = 0; kk < 8; ++kk) {
    const int k = k0 + kk * 4;
    Bs[n * ASTR + k]       = (_Float16)bv[kk].x;
    Bs[(n + 1) * ASTR + k] = (_Float16)bv[kk].y;
  }
}
__device__ inline void load_b_rowmajor(const float* __restrict__ W, int ldw,
                                       int n0, int kb, int tid, float2 bv[8]) {
  const int k0 = tid >> 6, n = (tid & 63) * 2;
#pragma unroll
  for (int kk = 0; kk < 8; ++kk) {
    const int k = k0 + kk * 4;
    bv[kk] = *(const float2*)(W + (kb + k) * ldw + n0 + n);
  }
}
__device__ inline void compute_tile(const _Float16* As, const _Float16* Bs,
                                    int wm, int wn, int lane, v8f acc[4][2]) {
  v16h bf[2], af[4];
#pragma unroll
  for (int nt = 0; nt < 2; ++nt) bf[nt] = frag_b(Bs + (wn * 32 + nt * 16) * ASTR, lane);
#pragma unroll
  for (int mt = 0; mt < 4; ++mt) af[mt] = frag_a(As + (wm * 64 + mt * 16) * ASTR, lane);
#pragma unroll
  for (int mt = 0; mt < 4; ++mt)
#pragma unroll
    for (int nt = 0; nt < 2; ++nt)
      acc[mt][nt] = __builtin_amdgcn_wmma_f32_16x16x32_f16(
          false, af[mt], false, bf[nt], (short)0, acc[mt][nt], false, false);
}

// ===========================================================================
// Kernel 1: x_shift = (x_cat @ Wshift)*gate + x*(1-gate)   [M=32768,N=K=1024]
// Double-buffered LDS; next K-tile prefetched into registers during WMMAs.
// ===========================================================================
__global__ __launch_bounds__(256)
void gemm_shift(const float* __restrict__ x, const float* __restrict__ Wshift,
                const float* __restrict__ shift_gate, float* __restrict__ xshift) {
  __shared__ _Float16 As[2][TSZ];
  __shared__ _Float16 Bs[2][TSZ];
  const int tid  = threadIdx.x;
  const int lane = tid & 31, wid = tid >> 5;
  const int wm = wid >> 2, wn = wid & 3;
  const int m0 = blockIdx.y * 128, n0 = blockIdx.x * 128;

  // per-thread x_cat source rows (kb-independent)
  const int rl = tid >> 3, kc = (tid & 7) * 4;
  int srcrow[4];
#pragma unroll
  for (int r4 = 0; r4 < 4; ++r4) {
    const int m = m0 + r4 * 32 + rl;
    const int b = m >> 13, t = m & (TLEN - 1);
    srcrow[r4] = (b << 13) | ((t + TLEN / 2) & (TLEN - 1));
  }

  v8f acc[4][2];
  const v8f vzero = {};
#pragma unroll
  for (int i = 0; i < 4; ++i)
#pragma unroll
    for (int j = 0; j < 2; ++j) acc[i][j] = vzero;

  float4 av[4]; float2 bv[8];
#pragma unroll
  for (int r4 = 0; r4 < 4; ++r4)
    av[r4] = *(const float4*)(x + srcrow[r4] * D_ATTN + kc);
  load_b_rowmajor(Wshift, D_ATTN, n0, 0, tid, bv);
  store_a_tile(As[0], tid, av);
  store_b_tile(Bs[0], tid, bv);
  __syncthreads();

  const int NSTEP = D_ATTN / AK;
  for (int s = 0; s < NSTEP; ++s) {
    const int cur = s & 1, nxt = cur ^ 1;
    const bool more = (s + 1 < NSTEP);
    if (more) {
      const int kb = (s + 1) * AK;
#pragma unroll
      for (int r4 = 0; r4 < 4; ++r4)
        av[r4] = *(const float4*)(x + srcrow[r4] * D_ATTN + kb + kc);
      load_b_rowmajor(Wshift, D_ATTN, n0, kb, tid, bv);
    }
    compute_tile(As[cur], Bs[cur], wm, wn, lane, acc);
    if (more) {
      store_a_tile(As[nxt], tid, av);
      store_b_tile(Bs[nxt], tid, bv);
    }
    __syncthreads();
  }

  // epilogue: y*g + x*(1-g)
  const int nI = lane & 15, hiI = lane >> 4;
#pragma unroll
  for (int nt = 0; nt < 2; ++nt) {
    const int n = n0 + wn * 32 + nt * 16 + nI;
    const float g = fast_sigmoid(shift_gate[n]);
#pragma unroll
    for (int mt = 0; mt < 4; ++mt) {
      const int mb = m0 + wm * 64 + mt * 16 + hiI * 8;
#pragma unroll
      for (int r = 0; r < 8; ++r) {
        const int m = mb + r;
        const float xv = x[m * D_ATTN + n];
        xshift[m * D_ATTN + n] = acc[mt][nt][r] * g + xv * (1.f - g);
      }
    }
  }
}

// ===========================================================================
// Kernel 2: per-row mean / rsqrt(var+eps) over 1024 elements
// ===========================================================================
__global__ __launch_bounds__(256)
void ln_stats(const float* __restrict__ xshift, float* __restrict__ mu_rstd) {
  const int tid = threadIdx.x;
  const int lane = tid & 31, wid = tid >> 5;
  const int row = blockIdx.x * 8 + wid;
  const float* p = xshift + row * D_ATTN;
  float s = 0.f, s2 = 0.f;
#pragma unroll
  for (int i = 0; i < 32; ++i) {
    const float v = p[lane + i * 32];
    s += v; s2 += v * v;
  }
#pragma unroll
  for (int off = 16; off > 0; off >>= 1) {
    s  += __shfl_xor(s,  off, 32);
    s2 += __shfl_xor(s2, off, 32);
  }
  if (lane == 0) {
    const float mu  = s  * (1.f / 1024.f);
    const float var = s2 * (1.f / 1024.f) - mu * mu;
    mu_rstd[row * 2]     = mu;
    mu_rstd[row * 2 + 1] = rsqrtf(var + 1e-5f);
  }
}

// ===========================================================================
// Kernel 3: kv = LN(x_shift) @ [Wk|Wv]      [M=32768, N=128, K=1024]
// LN fused into A staging; same double-buffered pipeline.
// ===========================================================================
__global__ __launch_bounds__(256)
void gemm_kv(const float* __restrict__ xshift, const float* __restrict__ Wk,
             const float* __restrict__ Wv, const float* __restrict__ ln_g,
             const float* __restrict__ ln_b, const float* __restrict__ mu_rstd,
             float* __restrict__ kv) {
  __shared__ _Float16 As[2][TSZ];
  __shared__ _Float16 Bs[2][TSZ];
  const int tid  = threadIdx.x;
  const int lane = tid & 31, wid = tid >> 5;
  const int wm = wid >> 2, wn = wid & 3;
  const int m0 = blockIdx.x * 128;

  const int rl = tid >> 3, kc = (tid & 7) * 4;
  // per-thread row stats, kb-independent
  float2 st[4];
#pragma unroll
  for (int r4 = 0; r4 < 4; ++r4)
    st[r4] = *(const float2*)(mu_rstd + (m0 + r4 * 32 + rl) * 2);
  // Wk/Wv side select, kb-independent (n even -> n,n+1 same side)
  const int k0 = tid >> 6, nb = (tid & 63) * 2;
  const float* Wsel = (nb < 64) ? Wk : Wv;
  const int nn = (nb < 64) ? nb : nb - 64;

  v8f acc[4][2];
  const v8f vzero = {};
#pragma unroll
  for (int i = 0; i < 4; ++i)
#pragma unroll
    for (int j = 0; j < 2; ++j) acc[i][j] = vzero;

  float4 av[4]; float2 bv[8]; float4 gv, bbv;
  auto lda = [&](int kb) {
#pragma unroll
    for (int r4 = 0; r4 < 4; ++r4)
      av[r4] = *(const float4*)(xshift + (m0 + r4 * 32 + rl) * D_ATTN + kb + kc);
    gv  = *(const float4*)(ln_g + kb + kc);
    bbv = *(const float4*)(ln_b + kb + kc);
  };
  auto ldb = [&](int kb) {
#pragma unroll
    for (int kk = 0; kk < 8; ++kk)
      bv[kk] = *(const float2*)(Wsel + (kb + k0 + kk * 4) * 64 + nn);
  };
  auto sta_ln = [&](_Float16* A) {
#pragma unroll
    for (int r4 = 0; r4 < 4; ++r4) {
      _Float16* dst = A + (r4 * 32 + rl) * ASTR + kc;
      dst[0] = (_Float16)((av[r4].x - st[r4].x) * st[r4].y * gv.x + bbv.x);
      dst[1] = (_Float16)((av[r4].y - st[r4].x) * st[r4].y * gv.y + bbv.y);
      dst[2] = (_Float16)((av[r4].z - st[r4].x) * st[r4].y * gv.z + bbv.z);
      dst[3] = (_Float16)((av[r4].w - st[r4].x) * st[r4].y * gv.w + bbv.w);
    }
  };

  lda(0); ldb(0);
  sta_ln(As[0]); store_b_tile(Bs[0], tid, bv);
  __syncthreads();

  const int NSTEP = D_ATTN / AK;
  for (int s = 0; s < NSTEP; ++s) {
    const int cur = s & 1, nxt = cur ^ 1;
    const bool more = (s + 1 < NSTEP);
    if (more) { lda((s + 1) * AK); ldb((s + 1) * AK); }
    compute_tile(As[cur], Bs[cur], wm, wn, lane, acc);
    if (more) { sta_ln(As[nxt]); store_b_tile(Bs[nxt], tid, bv); }
    __syncthreads();
  }

  const int nI = lane & 15, hiI = lane >> 4;
#pragma unroll
  for (int nt = 0; nt < 2; ++nt) {
    const int n = wn * 32 + nt * 16 + nI;
#pragma unroll
    for (int mt = 0; mt < 4; ++mt) {
      const int mb = m0 + wm * 64 + mt * 16 + hiI * 8;
#pragma unroll
      for (int r = 0; r < 8; ++r) kv[(mb + r) * 128 + n] = acc[mt][nt][r];
    }
  }
}

// ===========================================================================
// Kernel 4: wkv = exp(-exp(tf)*sigmoid(k))*v ; per-chunk EMA h = h*w + wkv
// ===========================================================================
__global__ __launch_bounds__(256)
void wkv_hs(const float* __restrict__ kv, const float* __restrict__ time_decay,
            const float* __restrict__ time_first, float* __restrict__ wkv,
            float* __restrict__ hs) {
  const int tid = threadIdx.x;
  const int d = tid & 63;
  const int chunk = blockIdx.x * 4 + (tid >> 6);
  const float w  = __expf(time_decay[d]);
  const float tf = __expf(time_first[d]);
  const int base = chunk * CHUNKL;
  float h = 0.f;
  for (int t = 0; t < CHUNKL; ++t) {
    const int m = base + t;
    const float kk = kv[m * 128 + d];
    const float vv = kv[m * 128 + 64 + d];
    const float u = __expf(-tf * fast_sigmoid(kk)) * vv;
    wkv[m * D_STATE + d] = u;
    h = h * w + u;
    hs[m * D_STATE + d] = h;
  }
}

// ===========================================================================
// Kernel 5: faithful intra-chunk stride scan (127 sequential passes) in LDS;
// emits only the last row per chunk (all the reference ever consumes).
// ===========================================================================
__global__ __launch_bounds__(256)
void intra_scan(const float* __restrict__ wkv, const float* __restrict__ time_decay,
                float* __restrict__ chunkst) {
  __shared__ float buf[CHUNKL * D_STATE];  // 32 KB
  const int tid = threadIdx.x;
  const int d = tid & 63, t6 = tid >> 6;   // element (i,d): i = 4q + t6
  const int chunk = blockIdx.x;
  const float w = __expf(time_decay[d]);
  float own[32];
#pragma unroll
  for (int q = 0; q < 32; ++q) {
    const int i = q * 4 + t6;
    own[q] = wkv[(chunk * CHUNKL + i) * D_STATE + d];
    buf[i * D_STATE + d] = own[q];
  }
  __syncthreads();
  for (int s = 1; s < CHUNKL; ++s) {
    float dep[32];
#pragma unroll
    for (int q = 0; q < 32; ++q) {
      const int i = q * 4 + t6;
      dep[q] = (i >= s) ? buf[(i - s) * D_STATE + d] : 0.f;
    }
    __syncthreads();
#pragma unroll
    for (int q = 0; q < 32; ++q) {
      const int i = q * 4 + t6;
      own[q] += w * dep[q];
      buf[i * D_STATE + d] = own[q];
    }
    __syncthreads();
  }
  if (t6 == 3) chunkst[chunk * D_STATE + d] = own[31];  // i == 127
}

// ===========================================================================
// Kernel 6: inter-chunk stride scan (S=64, w^128); writes last_state (B x 64)
// ===========================================================================
__global__ __launch_bounds__(256)
void inter_scan(const float* __restrict__ chunkst, const float* __restrict__ time_decay,
                float* __restrict__ last_state) {
  __shared__ float buf[CPB * D_STATE];     // 16 KB
  const int tid = threadIdx.x;
  const int d = tid & 63, t6 = tid >> 6;
  const int b = blockIdx.x;
  const float w = __expf(time_decay[d] * (float)CHUNKL);   // w^CHUNK
  float own[16];
#pragma unroll
  for (int q = 0; q < 16; ++q) {
    const int i = q * 4 + t6;
    own[q] = chunkst[(b * CPB + i) * D_STATE + d];
    buf[i * D_STATE + d] = own[q];
  }
  __syncthreads();
  for (int s = 1; s < CPB; ++s) {
    float dep[16];
#pragma unroll
    for (int q = 0; q < 16; ++q) {
      const int i = q * 4 + t6;
      dep[q] = (i >= s) ? buf[(i - s) * D_STATE + d] : 0.f;
    }
    __syncthreads();
#pragma unroll
    for (int q = 0; q < 16; ++q) {
      const int i = q * 4 + t6;
      own[q] += w * dep[q];
      buf[i * D_STATE + d] = own[q];
    }
    __syncthreads();
  }
  if (t6 == 3) last_state[b * D_STATE + d] = own[15];  // i == 63
}

// ===========================================================================
// Kernel 7: out = hs @ Wo       [M=32768, N=1024, K=64]
// ===========================================================================
__global__ __launch_bounds__(256)
void gemm_out(const float* __restrict__ hs, const float* __restrict__ Wo,
              float* __restrict__ out) {
  __shared__ _Float16 As[2][TSZ];
  __shared__ _Float16 Bs[2][TSZ];
  const int tid  = threadIdx.x;
  const int lane = tid & 31, wid = tid >> 5;
  const int wm = wid >> 2, wn = wid & 3;
  const int m0 = blockIdx.y * 128, n0 = blockIdx.x * 128;
  const int rl = tid >> 3, kc = (tid & 7) * 4;

  v8f acc[4][2];
  const v8f vzero = {};
#pragma unroll
  for (int i = 0; i < 4; ++i)
#pragma unroll
    for (int j = 0; j < 2; ++j) acc[i][j] = vzero;

  float4 av[4]; float2 bv[8];
  auto lda = [&](int kb) {
#pragma unroll
    for (int r4 = 0; r4 < 4; ++r4)
      av[r4] = *(const float4*)(hs + (m0 + r4 * 32 + rl) * D_STATE + kb + kc);
  };
  lda(0); load_b_rowmajor(Wo, D_ATTN, n0, 0, tid, bv);
  store_a_tile(As[0], tid, av);
  store_b_tile(Bs[0], tid, bv);
  __syncthreads();

  const int NSTEP = D_STATE / AK;  // 2
  for (int s = 0; s < NSTEP; ++s) {
    const int cur = s & 1, nxt = cur ^ 1;
    const bool more = (s + 1 < NSTEP);
    if (more) { lda((s + 1) * AK); load_b_rowmajor(Wo, D_ATTN, n0, (s + 1) * AK, tid, bv); }
    compute_tile(As[cur], Bs[cur], wm, wn, lane, acc);
    if (more) { store_a_tile(As[nxt], tid, av); store_b_tile(Bs[nxt], tid, bv); }
    __syncthreads();
  }

  const int nI = lane & 15, hiI = lane >> 4;
#pragma unroll
  for (int nt = 0; nt < 2; ++nt) {
    const int n = n0 + wn * 32 + nt * 16 + nI;
#pragma unroll
    for (int mt = 0; mt < 4; ++mt) {
      const int mb = m0 + wm * 64 + mt * 16 + hiI * 8;
#pragma unroll
      for (int r = 0; r < 8; ++r) out[(mb + r) * D_ATTN + n] = acc[mt][nt][r];
    }
  }
}

// ===========================================================================
extern "C" void kernel_launch(void* const* d_in, const int* in_sizes, int n_in,
                              void* d_out, int out_size, void* d_ws, size_t ws_size,
                              hipStream_t stream) {
  (void)in_sizes; (void)n_in; (void)out_size; (void)ws_size;
  const float* x          = (const float*)d_in[0];
  const float* time_decay = (const float*)d_in[1];
  const float* time_first = (const float*)d_in[2];
  const float* Wk         = (const float*)d_in[3];
  const float* Wv         = (const float*)d_in[4];
  const float* Wo         = (const float*)d_in[5];
  const float* Wshift     = (const float*)d_in[6];
  const float* shift_gate = (const float*)d_in[7];
  const float* ln_g       = (const float*)d_in[8];
  const float* ln_b       = (const float*)d_in[9];
  float* out = (float*)d_out;

  char* ws = (char*)d_ws;
  float* mu_rstd = (float*)(ws);                                  // 256 KB
  float* kv      = (float*)(ws + (1u << 20));                     // 16 MB
  float* wkvbuf  = (float*)(ws + (1u << 20) + (16u << 20));       // 8 MB
  float* hs      = (float*)(ws + (1u << 20) + (24u << 20));       // 8 MB
  float* chunkst = (float*)(ws + (1u << 20) + (32u << 20));       // 64 KB

  float* xshift     = out;                       // reuse d_out as scratch
  float* last_state = out + (size_t)MROWS * D_ATTN;

  gemm_shift<<<dim3(D_ATTN / 128, MROWS / 128), 256, 0, stream>>>(x, Wshift, shift_gate, xshift);
  ln_stats  <<<MROWS / 8, 256, 0, stream>>>(xshift, mu_rstd);
  gemm_kv   <<<MROWS / 128, 256, 0, stream>>>(xshift, Wk, Wv, ln_g, ln_b, mu_rstd, kv);
  wkv_hs    <<<NCHUNK / 4, 256, 0, stream>>>(kv, time_decay, time_first, wkvbuf, hs);
  intra_scan<<<NCHUNK, 256, 0, stream>>>(wkvbuf, time_decay, chunkst);
  inter_scan<<<BATCH, 256, 0, stream>>>(chunkst, time_decay, last_state);
  gemm_out  <<<dim3(D_ATTN / 128, MROWS / 128), 256, 0, stream>>>(hs, Wo, out);
}
